// HGRNAttention_30743375905045
// MI455X (gfx1250) — matrix operand order repaired and checked by reading, not compiled, not run on hardware
//
#include <hip/hip_runtime.h>
#include <cstdint>

// ---------------- problem constants (fixed by reference) ----------------
#define BB 4
#define LL 4096
#define DD 2048
#define MM (BB * LL)        // 16384 rows
#define NTOK (MM * DD)      // 33,554,432 elements per [B,L,D] tensor
#define NC 64               // scan chunks
#define CHUNK (LL / NC)     // 64 steps per chunk
#define CH (BB * DD)        // 8192 independent scan channels
#define EPS 1e-5f

typedef unsigned short u16;
typedef __attribute__((ext_vector_type(16))) __bf16 v16bf;
typedef __attribute__((ext_vector_type(8)))  __bf16 v8bf;
typedef __attribute__((ext_vector_type(8)))  float  v8f;

union bfrag { v16bf v; v8bf h[2]; };

__device__ __forceinline__ u16 f32_to_bf16(float f) {
  uint32_t u = __float_as_uint(f);
  uint32_t r = u + 0x7fffu + ((u >> 16) & 1u);   // round-to-nearest-even
  return (u16)(r >> 16);
}
__device__ __forceinline__ float sigmoidf_(float x) {
  return 1.0f / (1.0f + __expf(-x));
}

// CDNA5 async global->LDS DMA (ASYNCcnt-tracked, no VGPR round trip)
__device__ __forceinline__ void async_g2l_b128(void* lds, const void* g) {
  unsigned l = (unsigned)(uintptr_t)lds;                 // low 32 bits = LDS offset
  unsigned long long ga = (unsigned long long)(uintptr_t)g;
  asm volatile("global_load_async_to_lds_b128 %0, %1, off"
               :: "v"(l), "v"(ga) : "memory");
}
__device__ __forceinline__ void wait_async0() {
  asm volatile("s_wait_asynccnt 0x0" ::: "memory");
}

// ---------------- f32 -> bf16 conversion ----------------
__global__ void cvt_f32_bf16(const float* __restrict__ in, u16* __restrict__ out, int n) {
  int i = blockIdx.x * blockDim.x + threadIdx.x;
  int stride = gridDim.x * blockDim.x;
  for (; i < n; i += stride) out[i] = f32_to_bf16(in[i]);
}

// ---------------- bf16 WMMA GEMM: C[M,N] = A[M,K] * W[N,K]^T ----------------
// block: 256 threads (8 waves), tile 128x128, K-step 32, double-buffered LDS
// staged via global_load_async_to_lds_b128.
__global__ __launch_bounds__(256)
void gemm_bf16_wmma(const u16* __restrict__ A, const u16* __restrict__ W,
                    float* __restrict__ C, int M, int N, int K) {
  __shared__ u16 As[2][128][40];  // pad 32->40 halfwords: 16B align kept, banks spread
  __shared__ u16 Bs[2][128][40];

  const int tid  = threadIdx.x;
  const int lane = tid & 31;
  const int wave = tid >> 5;
  const int wm   = wave & 1;    // 2 waves over M: 64 rows each
  const int wn   = wave >> 1;   // 4 waves over N: 32 cols each
  const int m0   = blockIdx.y * 128;
  const int n0   = blockIdx.x * 128;

  const int ldRow  = tid >> 1;          // 0..127
  const int ldHalf = (tid & 1) * 16;    // halfword offset 0 or 16

  v8f acc[4][2];
  #pragma unroll
  for (int mt = 0; mt < 4; ++mt)
    #pragma unroll
    for (int nt = 0; nt < 2; ++nt)
      #pragma unroll
      for (int e = 0; e < 8; ++e) acc[mt][nt][e] = 0.0f;

  const u16* aRow = A + (size_t)(m0 + ldRow) * K + ldHalf;
  const u16* bRow = W + (size_t)(n0 + ldRow) * K + ldHalf;

  // stage one 128x32 A-tile + 128x32 B-tile into LDS buffer `buf` (32B/thread each)
  auto stage = [&](int buf, int kb) {
    async_g2l_b128(&As[buf][ldRow][ldHalf],     aRow + kb);
    async_g2l_b128(&As[buf][ldRow][ldHalf + 8], aRow + kb + 8);
    async_g2l_b128(&Bs[buf][ldRow][ldHalf],     bRow + kb);
    async_g2l_b128(&Bs[buf][ldRow][ldHalf + 8], bRow + kb + 8);
  };

  const int nsteps = K / 32;
  stage(0, 0);
  wait_async0();
  __syncthreads();

  for (int s = 0; s < nsteps; ++s) {
    const int cur = s & 1;
    if (s + 1 < nsteps) stage(1 - cur, (s + 1) * 32);   // prefetch next tile via async DMA

    // A fragments 16x32: lane = M row; lanes<16 hold K 0..7 & 16..23, lanes>=16 hold 8..15 & 24..31
    bfrag fa[4];
    const int ak = (lane >> 4) * 8;
    #pragma unroll
    for (int mt = 0; mt < 4; ++mt) {
      const int r = wm * 64 + mt * 16 + (lane & 15);
      fa[mt].h[0] = *(const v8bf*)&As[cur][r][ak];
      fa[mt].h[1] = *(const v8bf*)&As[cur][r][ak + 16];
    }
    // B fragments 32x16: lane%16 = N col; lanes<16 hold K 0..7 & 16..23, lanes>=16 hold 8..15 & 24..31
    bfrag fb[2];
    const int bk = (lane >> 4) * 8;
    #pragma unroll
    for (int nt = 0; nt < 2; ++nt) {
      const int r = wn * 32 + nt * 16 + (lane & 15);
      fb[nt].h[0] = *(const v8bf*)&Bs[cur][r][bk];
      fb[nt].h[1] = *(const v8bf*)&Bs[cur][r][bk + 16];
    }
    #pragma unroll
    for (int mt = 0; mt < 4; ++mt)
      #pragma unroll
      for (int nt = 0; nt < 2; ++nt)
        acc[mt][nt] = __builtin_amdgcn_wmma_f32_16x16x32_bf16(
            false, fa[mt].v, false, fb[nt].v, (short)0, acc[mt][nt], false, false);

    wait_async0();     // next tile landed (each wave waits its own DMAs)
    __syncthreads();   // all waves done reading `cur` and done landing `1-cur`
  }

  // writeback: VGPR j -> M = j + (lane>=16 ? 8 : 0), N = lane%16
  const int rbase = m0 + wm * 64 + (lane >> 4) * 8;
  const int cbase = n0 + wn * 32 + (lane & 15);
  #pragma unroll
  for (int mt = 0; mt < 4; ++mt)
    #pragma unroll
    for (int nt = 0; nt < 2; ++nt)
      #pragma unroll
      for (int j = 0; j < 8; ++j)
        C[(size_t)(rbase + mt * 16 + j) * N + (cbase + nt * 16)] = acc[mt][nt][j];
}

// ---------------- gating elementwise: gate=sigmoid(f); i_in=silu(i)*(1-gate) ----------------
__global__ void gate_ew(float* __restrict__ fbuf /* -> gate */,
                        float* __restrict__ ibuf /* -> i_in  */, int n) {
  int i = blockIdx.x * blockDim.x + threadIdx.x;
  int stride = gridDim.x * blockDim.x;
  for (; i < n; i += stride) {
    float fv = fbuf[i];
    float iv = ibuf[i];
    float gt = sigmoidf_(fv);
    ibuf[i] = iv * sigmoidf_(iv) * (1.0f - gt);
    fbuf[i] = gt;
  }
}

// ---------------- chunked associative scan: h_t = a_t*h_{t-1} + b_t ----------------
__global__ void scan_pass1(const float* __restrict__ gate, float* __restrict__ o,
                           float* __restrict__ aggA, float* __restrict__ aggB) {
  int t = blockIdx.x * blockDim.x + threadIdx.x;   // 0 .. CH*NC-1
  int ch = t % CH;                                 // b*D + d (coalesced in d)
  int c  = t / CH;
  int b  = ch / DD;
  int d  = ch % DD;
  size_t base = ((size_t)b * LL + (size_t)c * CHUNK) * DD + d;
  float h = 0.0f, p = 1.0f;
  for (int s = 0; s < CHUNK; ++s) {
    size_t idx = base + (size_t)s * DD;
    float a = gate[idx];
    float x = o[idx];
    h = a * h + x;
    p *= a;
    o[idx] = h;          // local scan in place
  }
  aggA[t] = p;           // layout [c][ch]
  aggB[t] = h;
}

__global__ void scan_pass2(const float* __restrict__ aggA, const float* __restrict__ aggB,
                           float* __restrict__ carry) {
  int ch = blockIdx.x * blockDim.x + threadIdx.x;  // 0..CH-1
  float h = 0.0f;
  for (int c = 0; c < NC; ++c) {
    int idx = c * CH + ch;
    carry[idx] = h;                                // exclusive carry into chunk c
    h = aggA[idx] * h + aggB[idx];
  }
}

__global__ void scan_pass3(const float* __restrict__ gate, float* __restrict__ o,
                           const float* __restrict__ carry) {
  int t = blockIdx.x * blockDim.x + threadIdx.x;
  int ch = t % CH;
  int c  = t / CH;
  float cin = carry[t];
  if (cin == 0.0f) return;                         // chunk 0 (and lucky zeros) skip
  int b = ch / DD;
  int d = ch % DD;
  size_t base = ((size_t)b * LL + (size_t)c * CHUNK) * DD + d;
  float p = 1.0f;
  for (int s = 0; s < CHUNK; ++s) {
    size_t idx = base + (size_t)s * DD;
    p *= gate[idx];
    o[idx] += cin * p;
  }
}

// ---------------- fused RMSNorm * weight * swish(o), output bf16 ----------------
__global__ __launch_bounds__(256)
void rms_swish_gate(const float* __restrict__ g, const float* __restrict__ o,
                    const float* __restrict__ w, u16* __restrict__ u) {
  __shared__ float red[256];
  const int row = blockIdx.x;          // 0..MM-1
  const int tid = threadIdx.x;
  size_t base = (size_t)row * DD;
  float ss = 0.0f;
  #pragma unroll
  for (int k = 0; k < DD / 256; ++k) {
    float v = g[base + tid + k * 256];
    ss += v * v;
  }
  red[tid] = ss;
  __syncthreads();
  for (int s = 128; s > 0; s >>= 1) {
    if (tid < s) red[tid] += red[tid + s];
    __syncthreads();
  }
  const float rms = rsqrtf(red[0] * (1.0f / DD) + EPS);
  #pragma unroll
  for (int k = 0; k < DD / 256; ++k) {
    int col = tid + k * 256;
    size_t idx = base + col;
    float oo = o[idx];
    float val = g[idx] * rms * w[col] * (oo * sigmoidf_(oo));
    u[idx] = f32_to_bf16(val);
  }
}

// ---------------- host-side orchestration ----------------
extern "C" void kernel_launch(void* const* d_in, const int* in_sizes, int n_in,
                              void* d_out, int out_size, void* d_ws, size_t ws_size,
                              hipStream_t stream) {
  const float* x   = (const float*)d_in[0];
  const float* Wi  = (const float*)d_in[1];
  const float* Wf  = (const float*)d_in[2];
  const float* Wg  = (const float*)d_in[3];
  const float* wgn = (const float*)d_in[4];
  const float* Wo  = (const float*)d_in[5];
  float* out = (float*)d_out;

  // workspace carve-up
  uint8_t* p = (uint8_t*)d_ws;
  u16* xh  = (u16*)p;                p += (size_t)NTOK * 2;            // 64 MiB (reused for u)
  u16* wih = (u16*)p;                p += (size_t)DD * DD * 2;
  u16* wfh = (u16*)p;                p += (size_t)DD * DD * 2;
  u16* wgh = (u16*)p;                p += (size_t)DD * DD * 2;
  u16* woh = (u16*)p;                p += (size_t)DD * DD * 2;         // +32 MiB
  float* iF = (float*)p;             p += (size_t)NTOK * 4;            // i -> i_in -> o
  float* fF = (float*)p;             p += (size_t)NTOK * 4;            // f -> gate
  float* gF = (float*)p;             p += (size_t)NTOK * 4;            // g
  float* aggA  = (float*)p;          p += (size_t)CH * NC * 4;
  float* aggB  = (float*)p;          p += (size_t)CH * NC * 4;
  float* carry = (float*)p;          p += (size_t)CH * NC * 4;
  u16* uh = xh;                                                        // alias: x done after GEMMs

  const dim3 blk(256);
  // 1) precision conversion
  cvt_f32_bf16<<<8192, blk, 0, stream>>>(x,  xh,  NTOK);
  cvt_f32_bf16<<<2048, blk, 0, stream>>>(Wi, wih, DD * DD);
  cvt_f32_bf16<<<2048, blk, 0, stream>>>(Wf, wfh, DD * DD);
  cvt_f32_bf16<<<2048, blk, 0, stream>>>(Wg, wgh, DD * DD);
  cvt_f32_bf16<<<2048, blk, 0, stream>>>(Wo, woh, DD * DD);

  // 2) three input projections via bf16 WMMA (f32 accum)
  const dim3 ggrid(DD / 128, MM / 128);   // (16, 128)
  gemm_bf16_wmma<<<ggrid, blk, 0, stream>>>(xh, wih, iF, MM, DD, DD);
  gemm_bf16_wmma<<<ggrid, blk, 0, stream>>>(xh, wfh, fF, MM, DD, DD);
  gemm_bf16_wmma<<<ggrid, blk, 0, stream>>>(xh, wgh, gF, MM, DD, DD);

  // 3) gating elementwise (in place)
  gate_ew<<<8192, blk, 0, stream>>>(fF, iF, NTOK);

  // 4) chunked linear-recurrence scan over L (in place in iF -> o)
  scan_pass1<<<(CH * NC) / 256, blk, 0, stream>>>(fF, iF, aggA, aggB);
  scan_pass2<<<CH / 256, blk, 0, stream>>>(aggA, aggB, carry);
  scan_pass3<<<(CH * NC) / 256, blk, 0, stream>>>(fF, iF, carry);

  // 5) fused RMSNorm * weight * swish(o) -> bf16
  rms_swish_gate<<<MM, blk, 0, stream>>>(gF, iF, wgn, uh);

  // 6) output projection
  gemm_bf16_wmma<<<ggrid, blk, 0, stream>>>(uh, woh, out, MM, DD, DD);
}
